// RobustGNN_37297495998500
// MI455X (gfx1250) — compile-verified, matrix operand order checked by reading, not run on hardware
//
#include <hip/hip_runtime.h>
#include <hip/hip_bf16.h>

#define N_NODES  16384
#define N_EDGES  262144
#define HID      128
#define OUT_DIM  2
#define MAX_DEG  48
#define TCAP     6

typedef __attribute__((ext_vector_type(16))) __bf16 v16bf;
typedef __attribute__((ext_vector_type(8)))  float  v8f;

static __device__ __forceinline__ __bf16 f2bf(float f) {
    // round-to-nearest-even fp32 -> bf16
    unsigned u = __builtin_bit_cast(unsigned, f);
    unsigned short h = (unsigned short)((u + 0x7FFFu + ((u >> 16) & 1u)) >> 16);
    return __builtin_bit_cast(__bf16, h);
}

// ---------------------------------------------------------------------------
// CSR build: cnt[v] counts in-degree (incl. self loop), adj stores up to
// MAX_DEG source ids per destination. Then a deterministic per-node sort.
// ---------------------------------------------------------------------------
__global__ void init_adj_kernel(int* __restrict__ cnt, int* __restrict__ adj) {
    int v = blockIdx.x * blockDim.x + threadIdx.x;
    if (v >= N_NODES) return;
    cnt[v] = 1;                 // self loop
    adj[v * MAX_DEG] = v;
}

__global__ void build_adj_kernel(const int* __restrict__ ei,
                                 int* __restrict__ cnt, int* __restrict__ adj) {
    int e = blockIdx.x * blockDim.x + threadIdx.x;
    if (e >= N_EDGES) return;
    int src = ei[e];
    int dst = ei[N_EDGES + e];
    int slot = atomicAdd(&cnt[dst], 1);
    if (slot < MAX_DEG) adj[dst * MAX_DEG + slot] = src;
}

__global__ void sort_adj_kernel(const int* __restrict__ cnt, int* __restrict__ adj) {
    int v = blockIdx.x * blockDim.x + threadIdx.x;
    if (v >= N_NODES) return;
    int m = cnt[v]; if (m > MAX_DEG) m = MAX_DEG;
    int a[MAX_DEG];
    for (int i = 0; i < m; ++i) a[i] = adj[v * MAX_DEG + i];
    for (int i = 1; i < m; ++i) {                 // insertion sort -> determinism
        int key = a[i]; int j = i - 1;
        while (j >= 0 && a[j] > key) { a[j + 1] = a[j]; --j; }
        a[j + 1] = key;
    }
    for (int i = 0; i < m; ++i) adj[v * MAX_DEG + i] = a[i];
}

// ---------------------------------------------------------------------------
// Fused dual GEMM:  Yl = X@Wl + bl,  Yr = X@Wr + br   (K = NC = 128)
// One wave computes one 16x16 tile of both outputs; A tile loaded once,
// two v_wmma_f32_16x16x32_bf16 per K-chunk.
// A layout (16-bit, 16x32): lane L, m = L&15; lanes 0-15 hold K 0-7 & 16-23,
// lanes 16-31 hold K 8-15 & 24-31 (ISA 7.12.2). B mirrors with n = L&15.
// C/D layout: VGPR r holds row r + 8*(L>>4), col = L&15.
// ---------------------------------------------------------------------------
__global__ __launch_bounds__(32)
void gemm_dual_wmma_kernel(const float* __restrict__ X,
                           const float* __restrict__ Wl, const float* __restrict__ bl,
                           const float* __restrict__ Wr, const float* __restrict__ br,
                           float* __restrict__ Yl, float* __restrict__ Yr) {
    const int K = 128, NC = 128;
    const int m0   = blockIdx.x * 16;
    const int n0   = blockIdx.y * 16;
    const int lane = threadIdx.x & 31;
    const int ml   = lane & 15;
    const int half = lane >> 4;

    v8f accl, accr;
    const float bvl = bl[n0 + ml];
    const float bvr = br[n0 + ml];
#pragma unroll
    for (int r = 0; r < 8; ++r) { accl[r] = bvl; accr[r] = bvr; }

    const float* arow = X + (long)(m0 + ml) * K;
    for (int kk = 0; kk < K; kk += 32) {
        const int kb = kk + half * 8;
        v16bf a, b1, b2;
#pragma unroll
        for (int j = 0; j < 8; ++j) {
            const int k = kb + ((j < 4) ? 2 * j : 16 + 2 * (j - 4));
            const float2 av = *(const float2*)(arow + k);
            a[2 * j]     = f2bf(av.x);
            a[2 * j + 1] = f2bf(av.y);
            b1[2 * j]     = f2bf(Wl[(long)k       * NC + n0 + ml]);
            b1[2 * j + 1] = f2bf(Wl[(long)(k + 1) * NC + n0 + ml]);
            b2[2 * j]     = f2bf(Wr[(long)k       * NC + n0 + ml]);
            b2[2 * j + 1] = f2bf(Wr[(long)(k + 1) * NC + n0 + ml]);
        }
        accl = __builtin_amdgcn_wmma_f32_16x16x32_bf16(false, a, false, b1,
                                                       (short)0, accl, false, false);
        accr = __builtin_amdgcn_wmma_f32_16x16x32_bf16(false, a, false, b2,
                                                       (short)0, accr, false, false);
    }
#pragma unroll
    for (int r = 0; r < 8; ++r) {
        const long row = m0 + r + 8 * half;
        Yl[row * NC + n0 + ml] = accl[r];
        Yr[row * NC + n0 + ml] = accr[r];
    }
}

// ---------------------------------------------------------------------------
// Trimmed-mean aggregation (HID features): block = node, thread = feature.
// msg = L[src]/deg ; trimmed mean = (sum - t smallest - t largest)/(deg*tcnt);
// plain-mean fallback = sum/deg^2. Then + R, BN (eval), ReLU.
// Selection via running t-min/t-max trackers (no sort needed, t <= TCAP).
// ---------------------------------------------------------------------------
__global__ __launch_bounds__(HID)
void agg_bn_relu_kernel(const float* __restrict__ Lb, const float* __restrict__ Rb,
                        const int* __restrict__ cnt, const int* __restrict__ adj,
                        const float* __restrict__ g, const float* __restrict__ be,
                        float* __restrict__ outp) {
    __shared__ int s_adj[MAX_DEG];
    const int v = blockIdx.x;
    const int f = threadIdx.x;
    if (f < MAX_DEG) s_adj[f] = adj[v * MAX_DEG + f];
    __syncthreads();

    const int c = cnt[v];
    int m = c < MAX_DEG ? c : MAX_DEG;
    int t = (int)((float)c * 0.1f);
    if (t < 1) t = 1; if (t > TCAP) t = TCAP;
    const int tcnt = c - 2 * t;
    const bool use_trim = (c >= 5) && (tcnt > 0) && (2 * t < m);

    float lo[TCAP], hi[TCAP];
#pragma unroll
    for (int j = 0; j < TCAP; ++j) { lo[j] = __builtin_inff(); hi[j] = -__builtin_inff(); }
    float sum = 0.0f;
    for (int i = 0; i < m; ++i) {
        const int u = s_adj[i];
        float x = Lb[(long)u * HID + f];
        sum += x;
        float xl = x, xh = x;
#pragma unroll
        for (int j = 0; j < TCAP; ++j)
            if (xl < lo[j]) { float tmp = lo[j]; lo[j] = xl; xl = tmp; }
#pragma unroll
        for (int j = 0; j < TCAP; ++j)
            if (xh > hi[j]) { float tmp = hi[j]; hi[j] = xh; xh = tmp; }
    }
    const float invdeg = 1.0f / (float)c;
    float agg;
    if (use_trim) {
        float s = sum;
#pragma unroll
        for (int j = 0; j < TCAP; ++j) if (j < t) { s -= lo[j]; s -= hi[j]; }
        agg = s * invdeg / (float)tcnt;
    } else {
        agg = sum * invdeg * invdeg;
    }
    float r = agg + Rb[(long)v * HID + f];
    const float bn = 0.99999500003749968f;     // 1/sqrt(1 + 1e-5)
    r = r * bn * g[f] + be[f];
    outp[(long)v * HID + f] = r > 0.0f ? r : 0.0f;
}

// ---------------------------------------------------------------------------
// Final layer projection: L2 = H@Wl2 + bl2, R2 = H@Wr2 + br2 (OUT_DIM = 2).
// ---------------------------------------------------------------------------
__global__ void lin2_kernel(const float* __restrict__ H,
                            const float* __restrict__ Wl, const float* __restrict__ bl,
                            const float* __restrict__ Wr, const float* __restrict__ br,
                            float* __restrict__ L2, float* __restrict__ R2) {
    int v = blockIdx.x * blockDim.x + threadIdx.x;
    if (v >= N_NODES) return;
    const float* h = H + (long)v * HID;
    float l0 = bl[0], l1 = bl[1], r0 = br[0], r1 = br[1];
    for (int k = 0; k < HID; ++k) {
        const float x = h[k];
        l0 = fmaf(x, Wl[k * 2 + 0], l0);
        l1 = fmaf(x, Wl[k * 2 + 1], l1);
        r0 = fmaf(x, Wr[k * 2 + 0], r0);
        r1 = fmaf(x, Wr[k * 2 + 1], r1);
    }
    L2[v * 2 + 0] = l0; L2[v * 2 + 1] = l1;
    R2[v * 2 + 0] = r0; R2[v * 2 + 1] = r1;
}

__global__ void agg_final_kernel(const float* __restrict__ L2, const float* __restrict__ R2,
                                 const int* __restrict__ cnt, const int* __restrict__ adj,
                                 float* __restrict__ outp) {
    int v = blockIdx.x * blockDim.x + threadIdx.x;
    if (v >= N_NODES) return;
    const int c = cnt[v];
    int m = c < MAX_DEG ? c : MAX_DEG;
    int t = (int)((float)c * 0.1f);
    if (t < 1) t = 1; if (t > TCAP) t = TCAP;
    const int tcnt = c - 2 * t;
    const bool use_trim = (c >= 5) && (tcnt > 0) && (2 * t < m);

    float lo0[TCAP], hi0[TCAP], lo1[TCAP], hi1[TCAP];
#pragma unroll
    for (int j = 0; j < TCAP; ++j) {
        lo0[j] = __builtin_inff(); hi0[j] = -__builtin_inff();
        lo1[j] = __builtin_inff(); hi1[j] = -__builtin_inff();
    }
    float s0 = 0.0f, s1 = 0.0f;
    for (int i = 0; i < m; ++i) {
        const int u = adj[v * MAX_DEG + i];
        float x0 = L2[(long)u * 2 + 0];
        float x1 = L2[(long)u * 2 + 1];
        s0 += x0; s1 += x1;
        float a = x0, b = x0;
#pragma unroll
        for (int j = 0; j < TCAP; ++j)
            if (a < lo0[j]) { float tmp = lo0[j]; lo0[j] = a; a = tmp; }
#pragma unroll
        for (int j = 0; j < TCAP; ++j)
            if (b > hi0[j]) { float tmp = hi0[j]; hi0[j] = b; b = tmp; }
        a = x1; b = x1;
#pragma unroll
        for (int j = 0; j < TCAP; ++j)
            if (a < lo1[j]) { float tmp = lo1[j]; lo1[j] = a; a = tmp; }
#pragma unroll
        for (int j = 0; j < TCAP; ++j)
            if (b > hi1[j]) { float tmp = hi1[j]; hi1[j] = b; b = tmp; }
    }
    const float invdeg = 1.0f / (float)c;
    float a0, a1;
    if (use_trim) {
        float u0 = s0, u1 = s1;
#pragma unroll
        for (int j = 0; j < TCAP; ++j)
            if (j < t) { u0 -= lo0[j] + hi0[j]; u1 -= lo1[j] + hi1[j]; }
        const float sc = invdeg / (float)tcnt;
        a0 = u0 * sc; a1 = u1 * sc;
    } else {
        a0 = s0 * invdeg * invdeg; a1 = s1 * invdeg * invdeg;
    }
    outp[v * 2 + 0] = a0 + R2[v * 2 + 0];
    outp[v * 2 + 1] = a1 + R2[v * 2 + 1];
}

// ---------------------------------------------------------------------------
extern "C" void kernel_launch(void* const* d_in, const int* in_sizes, int n_in,
                              void* d_out, int out_size, void* d_ws, size_t ws_size,
                              hipStream_t stream) {
    (void)in_sizes; (void)n_in; (void)out_size; (void)ws_size;
    const float* x   = (const float*)d_in[0];
    const int*   ei  = (const int*)  d_in[1];
    const float* Wl0 = (const float*)d_in[2];  const float* bl0 = (const float*)d_in[3];
    const float* Wr0 = (const float*)d_in[4];  const float* br0 = (const float*)d_in[5];
    const float* g0  = (const float*)d_in[6];  const float* be0 = (const float*)d_in[7];
    const float* Wl1 = (const float*)d_in[8];  const float* bl1 = (const float*)d_in[9];
    const float* Wr1 = (const float*)d_in[10]; const float* br1 = (const float*)d_in[11];
    const float* g1  = (const float*)d_in[12]; const float* be1 = (const float*)d_in[13];
    const float* Wl2 = (const float*)d_in[14]; const float* bl2 = (const float*)d_in[15];
    const float* Wr2 = (const float*)d_in[16]; const float* br2 = (const float*)d_in[17];
    float* out = (float*)d_out;

    char* ws = (char*)d_ws;
    int*   cnt  = (int*)ws;                                   ws += (size_t)N_NODES * 4;
    int*   adj  = (int*)ws;                                   ws += (size_t)N_NODES * MAX_DEG * 4;
    float* Lbuf = (float*)ws;                                 ws += (size_t)N_NODES * HID * 4;
    float* Rbuf = (float*)ws;                                 ws += (size_t)N_NODES * HID * 4;
    float* Hbuf = (float*)ws;                                 ws += (size_t)N_NODES * HID * 4;
    float* L2   = (float*)ws;                                 ws += (size_t)N_NODES * OUT_DIM * 4;
    float* R2   = (float*)ws;

    // CSR build (deterministic after per-node sort)
    init_adj_kernel <<<N_NODES / 256, 256, 0, stream>>>(cnt, adj);
    build_adj_kernel<<<N_EDGES / 256, 256, 0, stream>>>(ei, cnt, adj);
    sort_adj_kernel <<<N_NODES / 256, 256, 0, stream>>>(cnt, adj);

    dim3 ggrid(N_NODES / 16, HID / 16);

    // Layer 0
    gemm_dual_wmma_kernel<<<ggrid, 32, 0, stream>>>(x, Wl0, bl0, Wr0, br0, Lbuf, Rbuf);
    agg_bn_relu_kernel<<<N_NODES, HID, 0, stream>>>(Lbuf, Rbuf, cnt, adj, g0, be0, Hbuf);

    // Layer 1
    gemm_dual_wmma_kernel<<<ggrid, 32, 0, stream>>>(Hbuf, Wl1, bl1, Wr1, br1, Lbuf, Rbuf);
    agg_bn_relu_kernel<<<N_NODES, HID, 0, stream>>>(Lbuf, Rbuf, cnt, adj, g1, be1, Hbuf);

    // Layer 2 (HID -> 2) + final aggregation
    lin2_kernel     <<<N_NODES / 256, 256, 0, stream>>>(Hbuf, Wl2, bl2, Wr2, br2, L2, R2);
    agg_final_kernel<<<N_NODES / 256, 256, 0, stream>>>(L2, R2, cnt, adj, out);
}
